// FPQuantLinear_29042568855820
// MI455X (gfx1250) — 8-bit matrix operand order repaired and checked by reading, compiled, not run on hardware
//
#include <hip/hip_runtime.h>
#include <hip/hip_bf16.h>
#include <stdint.h>


typedef int   v16i __attribute__((ext_vector_type(16)));
typedef float v8f  __attribute__((ext_vector_type(8)));

#define GROUP 32

__device__ __forceinline__ unsigned char enc_e4m3(float f) {
  unsigned u    = __float_as_uint(f);
  unsigned sgn  = (u >> 31) << 7;
  unsigned absu = u & 0x7fffffffu;
  if (absu == 0u) return (unsigned char)sgn;
  int      expo = (int)((absu >> 23) & 0xffu) - 127;
  unsigned mant = (absu >> 20) & 0x7u;
  int      be   = expo + 7;
  if (be >= 16) return (unsigned char)(sgn | 0x7Eu);
  if (be <= 0) {
    int sh = 1 - be;
    unsigned m = (sh > 4) ? 0u : ((8u + mant) >> sh);
    if (m > 7u) m = 7u;
    return (unsigned char)(sgn | m);
  }
  return (unsigned char)(sgn | ((unsigned)be << 3) | mant);
}

__global__ __launch_bounds__(256) void rotate_quant_fp8(
    const float* __restrict__ in, const float* __restrict__ H,
    unsigned char* __restrict__ out, long long n_groups) {
  const int lane = threadIdx.x & 31;
  long long wave = (long long)blockIdx.x * (blockDim.x >> 5) + (threadIdx.x >> 5);
  if (wave >= n_groups) return;

  const float h00 = H[0];

  float v = __builtin_nontemporal_load(in + wave * GROUP + lane);

#pragma unroll
  for (int s = 0; s < 5; ++s) {
    float p = __shfl_xor(v, 1 << s, 32);
    v = (lane & (1 << s)) ? (p - v) : (v + p);
  }
  float acc = v * h00;

  float a = fabsf(acc);
#pragma unroll
  for (int off = 16; off > 0; off >>= 1)
    a = fmaxf(a, __shfl_xor(a, off, 32));
  a = fmaxf(a, 1.1754943508222875e-38f);

  int E;
  (void)frexpf(a, &E);
  int e = E - 3;
  e = (e > 127) ? 127 : ((e < -127) ? -127 : e);
  float scale = ldexpf(1.0f, e);

  float s2  = acc / scale;
  float mag = fminf(fabsf(s2), 6.0f);
  int idx = (mag > 0.25f) + (mag > 0.75f) + (mag > 1.25f) + (mag > 1.75f) +
            (mag > 2.5f)  + (mag > 3.5f)  + (mag > 5.0f);
  float g = (idx <= 4) ? 0.5f * (float)idx
                       : ((idx < 7) ? (float)(idx - 2) : 6.0f);
  float deq = copysignf(g, s2) * scale;

  out[wave * GROUP + lane] = enc_e4m3(deq);
}

#define BM 128
#define BN 64
#define BK 128
#define APITCH 144
#define BPITCH 144

#define ASYNC_LOAD_B128(ldsoff, gaddr)                                  \
  asm volatile("global_load_async_to_lds_b128 %0, %1, off"              \
               :: "v"(ldsoff), "v"(gaddr) : "memory")
#define WAIT_ASYNCCNT(n)                                                \
  asm volatile("s_wait_asynccnt " #n ::: "memory")

__global__ __launch_bounds__(256) void gemm_fp8_wmma(
    const unsigned char* __restrict__ Aq,
    const unsigned char* __restrict__ Bq,
    const float* __restrict__ bias,
    float* __restrict__ out,
    int M, int N, int K) {
  __shared__ unsigned char As[2][BM * APITCH];
  __shared__ unsigned char Bs[2][BN * BPITCH];

  const int tid  = threadIdx.x;
  const int lane = tid & 31;
  const int w    = tid >> 5;
  const int wm   = w & 3;
  const int wn   = w >> 2;
  const int h    = lane >> 4;
  const int l16  = lane & 15;

  const int m0 = blockIdx.y * BM;
  const int n0 = blockIdx.x * BN;

  const int arow0 = tid >> 3;
  const int akoff = (tid & 7) * 16;
  auto stage_async = [&](int k0, int buf) {
#pragma unroll
    for (int i = 0; i < 4; ++i) {
      int row = arow0 + 32 * i;
      unsigned lds = (unsigned)(uintptr_t)&As[buf][row * APITCH + akoff];
      unsigned long long g =
          (unsigned long long)(uintptr_t)(Aq + (size_t)(m0 + row) * K + k0 + akoff);
      ASYNC_LOAD_B128(lds, g);
    }
#pragma unroll
    for (int i = 0; i < 2; ++i) {
      int row = arow0 + 32 * i;
      unsigned lds = (unsigned)(uintptr_t)&Bs[buf][row * BPITCH + akoff];
      unsigned long long g =
          (unsigned long long)(uintptr_t)(Bq + (size_t)(n0 + row) * K + k0 + akoff);
      ASYNC_LOAD_B128(lds, g);
    }
  };

  v8f acc[2][2];
#pragma unroll
  for (int i = 0; i < 2; ++i)
#pragma unroll
    for (int j = 0; j < 2; ++j)
#pragma unroll
      for (int r = 0; r < 8; ++r) acc[i][j][r] = 0.0f;

  const int nk = K / BK;
  stage_async(0, 0);

  for (int t = 0; t < nk; ++t) {
    const int cur = t & 1;
    if (t + 1 < nk) {
      __syncthreads();
      stage_async((t + 1) * BK, cur ^ 1);
      WAIT_ASYNCCNT(6);
      if (t + 2 < nk)
        __builtin_prefetch(Aq + (size_t)(m0 + arow0) * K + (t + 2) * BK + akoff, 0, 1);
    } else {
      WAIT_ASYNCCNT(0);
    }
    __syncthreads();

    v16i afrag[2], bfrag[2];
#pragma unroll
    for (int sm = 0; sm < 2; ++sm) {
      const unsigned char* base =
          &As[cur][(wm * 32 + sm * 16 + l16) * APITCH + 16 * h];
#pragma unroll
      for (int j = 0; j < 8; ++j) {
        uint2 d = *(const uint2*)(base + 32 * (j >> 1) + 8 * (j & 1));
        afrag[sm][2 * j]     = (int)d.x;
        afrag[sm][2 * j + 1] = (int)d.y;
      }
    }
#pragma unroll
    for (int sn = 0; sn < 2; ++sn) {
      const unsigned char* base =
          &Bs[cur][(wn * 32 + sn * 16 + l16) * BPITCH + 16 * h];
#pragma unroll
      for (int j = 0; j < 4; ++j) {
        uint4 d = *(const uint4*)(base + 32 * j);
        bfrag[sn][4 * j]     = (int)d.x;
        bfrag[sn][4 * j + 1] = (int)d.y;
        bfrag[sn][4 * j + 2] = (int)d.z;
        bfrag[sn][4 * j + 3] = (int)d.w;
      }
    }

#pragma unroll
    for (int sm = 0; sm < 2; ++sm)
#pragma unroll
      for (int sn = 0; sn < 2; ++sn)
        acc[sm][sn] = __builtin_amdgcn_wmma_f32_16x16x128_fp8_fp8(
            afrag[sm], bfrag[sn], (short)0, acc[sm][sn], false, false);
  }

#pragma unroll
  for (int sm = 0; sm < 2; ++sm) {
#pragma unroll
    for (int sn = 0; sn < 2; ++sn) {
      int col = n0 + wn * 32 + sn * 16 + l16;
      float bv = bias[col];
#pragma unroll
      for (int r = 0; r < 8; ++r) {
        int rowm = m0 + wm * 32 + sm * 16 + r + 8 * h;
        __builtin_nontemporal_store(acc[sm][sn][r] + bv,
                                    out + (size_t)rowm * N + col);
      }
    }
  }
}

extern "C" void kernel_launch(void* const* d_in, const int* in_sizes, int n_in,
                              void* d_out, int out_size, void* d_ws, size_t ws_size,
                              hipStream_t stream) {
  const float* x    = (const float*)d_in[0];
  const float* wgt  = (const float*)d_in[1];
  const float* bias = (const float*)d_in[2];
  const float* H    = (const float*)d_in[3];

  const int N = in_sizes[2];
  const int K = in_sizes[1] / N;
  const int M = in_sizes[0] / K;

  unsigned char* xq = (unsigned char*)d_ws;
  unsigned char* wq = xq + (size_t)M * (size_t)K;

  const long long gx = (long long)M * K / GROUP;
  const long long gw = (long long)N * K / GROUP;

  rotate_quant_fp8<<<(unsigned)((gx + 7) / 8), 256, 0, stream>>>(x,   H, xq, gx);
  rotate_quant_fp8<<<(unsigned)((gw + 7) / 8), 256, 0, stream>>>(wgt, H, wq, gw);

  dim3 grid((unsigned)(N / BN), (unsigned)(M / BM));
  gemm_fp8_wmma<<<grid, 256, 0, stream>>>(xq, wq, bias, (float*)d_out, M, N, K);
}
